// STEncoderBlock_13331578487345
// MI455X (gfx1250) — compile-verified, matrix operand order checked by reading, not run on hardware
//
#include <hip/hip_runtime.h>

// ---------------- problem constants ----------------
#define TOKS   196608L     // B*T*N
#define DM     128
#define NNODES 2048
#define TSTEPS 24
#define NHEADS 4
#define HDIM   32
#define NEXP   8
#define MINTER 256
#define INTERD 512
#define ALPHA_MIX 0.05f

typedef __attribute__((ext_vector_type(16))) __bf16 v16bf;
typedef __attribute__((ext_vector_type(8)))  __bf16 v8bf;
typedef __attribute__((ext_vector_type(8)))  float  v8f;

__device__ __forceinline__ unsigned short f2bf_u(float f) {
  union { float f; unsigned u; } c; c.f = f;
  unsigned u = c.u + 0x7FFFu + ((c.u >> 16) & 1u);   // round-to-nearest-even
  return (unsigned short)(u >> 16);
}

// CDNA5 async direct copy global->LDS (16B), tracked by ASYNCcnt.
// LDS flat addresses: low 32 bits == LDS byte offset (aperture base low bits 0),
// so truncating the generic shared pointer gives the VDST operand.
__device__ __forceinline__ void async_ld16(void* ldsp, const void* gp) {
  asm volatile("global_load_async_to_lds_b128 %0, %1, off"
               :: "v"((unsigned)(unsigned long)ldsp),
                  "v"((unsigned long)gp)
               : "memory");
}
__device__ __forceinline__ void wait_async0() {
  asm volatile("s_wait_asynccnt 0" ::: "memory");
}

// Fragment loader following CDNA5 16-bit A/B VGPR layout (ISA 7.12.2):
// lane 0-15 -> row = lane, lane 16-31 -> row = lane-16 with K shifted by 8.
// elements 0..7  = K in [half*8, half*8+8)
// elements 8..15 = K in [16+half*8, 16+half*8+8)
__device__ __forceinline__ v16bf load_frag(const unsigned short* rowp, int hl) {
  v8bf lo = *(const v8bf*)(rowp + hl * 8);
  v8bf hi = *(const v8bf*)(rowp + hl * 8 + 16);
  return __builtin_shufflevector(lo, hi, 0,1,2,3,4,5,6,7,8,9,10,11,12,13,14,15);
}

// ---------------------------------------------------------------------------
// Generic WMMA GEMM:  C[M x N] = A[M x K] @ op(B)
//   TRANSB=false: B stored [N x K] row-major (torch Linear weight, C = A@B^T)
//   TRANSB=true : B stored [K x N] row-major (C = A@B), staged transposed
// Tiles: BM=64 BN=64 BK=32; 256 threads = 8 waves; each wave owns a 16x32
// output strip (2 accumulators, shared A fragment). Double-buffered LDS with
// async global->LDS copies overlapped with WMMA; one barrier per k-stage.
// Epilogue MODE:
//   0: outF = acc + bias[n]
//   1: outB = bf16(acc)
//   2: outB = bf16(alpha*resid + (1-alpha)*acc)          (GCN residual mix)
//   3: outF = resid + acc + bias[n]                      (block residual)
//   4: outF += rowscale[m*rsStride] * acc                (MoE scaled accum)
// ---------------------------------------------------------------------------
template <int MODE, bool TRANSB>
__global__ __launch_bounds__(256) void gemm_wmma(
    const unsigned short* __restrict__ A,
    const unsigned short* __restrict__ B,
    int M, int N, int K,
    long sAz, long sBz, long sCz,
    float* __restrict__ outF,
    unsigned short* __restrict__ outB,
    const float* __restrict__ bias,
    const float* __restrict__ resid,
    const float* __restrict__ rowscale,
    int rsStride, float alpha)
{
  (void)M;
  constexpr int LDK = 40;                      // shorts; 80B row stride, 16B aligned
  __shared__ __align__(16) unsigned short sA[2][64 * LDK];
  __shared__ __align__(16) unsigned short sB[2][64 * LDK];

  const int tid  = threadIdx.x;
  const int z    = blockIdx.z;
  const unsigned short* Ag = A + (long)z * sAz;
  const unsigned short* Bg = B + (long)z * sBz;
  const int m0 = blockIdx.x * 64;
  const int n0 = blockIdx.y * 64;
  const int wave = tid >> 5, lane = tid & 31;
  const int tm = (wave >> 1) * 16, tn = (wave & 1) * 32;
  const int hl = lane >> 4, ln = lane & 15;

  v8f accL = {}, accH = {};

  const int arow = tid >> 2, acol = (tid & 3) * 8;   // 64 rows x (4x8) cols

  auto stage = [&](int buf, int k0) {
    async_ld16(&sA[buf][arow * LDK + acol],
               &Ag[(long)(m0 + arow) * K + k0 + acol]);
    if constexpr (!TRANSB) {
      async_ld16(&sB[buf][arow * LDK + acol],
                 &Bg[(long)(n0 + arow) * K + k0 + acol]);
    } else {
      const int r = tid >> 3, c = (tid & 7) * 8;     // 32 k-rows x 64 n-cols
      const unsigned short* src = &Bg[(long)(k0 + r) * N + n0 + c];
#pragma unroll
      for (int i = 0; i < 8; ++i) sB[buf][(c + i) * LDK + r] = src[i];
    }
  };

  stage(0, 0);
  const int nk = K >> 5;
  for (int s = 0; s < nk; ++s) {
    wait_async0();
    __syncthreads();                       // publishes stage s; proves s-1 read done
    if (s + 1 < nk) stage((s + 1) & 1, (s + 1) << 5);
    const unsigned short* bufA = sA[s & 1];
    const unsigned short* bufB = sB[s & 1];
    v16bf a  = load_frag(&bufA[(tm + ln) * LDK], hl);
    v16bf bL = load_frag(&bufB[(tn + ln) * LDK], hl);
    v16bf bH = load_frag(&bufB[(tn + 16 + ln) * LDK], hl);
    accL = __builtin_amdgcn_wmma_f32_16x16x32_bf16(false, a, false, bL,
                                                   (short)0, accL, false, false);
    accH = __builtin_amdgcn_wmma_f32_16x16x32_bf16(false, a, false, bH,
                                                   (short)0, accH, false, false);
  }

  const long cz = (long)z * sCz;
#pragma unroll
  for (int t = 0; t < 2; ++t) {
    const v8f acc = t ? accH : accL;
#pragma unroll
    for (int r = 0; r < 8; ++r) {
      const int  gm  = m0 + tm + r + hl * 8;
      const int  gn  = n0 + tn + t * 16 + ln;
      const long idx = cz + (long)gm * N + gn;
      const float v  = acc[r];
      if constexpr (MODE == 0)      outF[idx] = v + (bias ? bias[gn] : 0.f);
      else if constexpr (MODE == 1) outB[idx] = f2bf_u(v);
      else if constexpr (MODE == 2) outB[idx] = f2bf_u(alpha * resid[idx] + (1.f - alpha) * v);
      else if constexpr (MODE == 3) outF[idx] = resid[idx] + v + bias[gn];
      else if constexpr (MODE == 4) outF[idx] += rowscale[(long)gm * rsStride] * v;
    }
  }
}

// Dual-B SWIGLU GEMM: out = bf16( silu(A@B1^T) * (A@B3^T) ), B1/B3 [N x K] rm.
// Same BM=64 BN=64 double-buffered async pipeline; 4 WMMAs per wave per stage.
__global__ __launch_bounds__(256) void gemm_wmma_swiglu(
    const unsigned short* __restrict__ A,
    const unsigned short* __restrict__ B1,
    const unsigned short* __restrict__ B3,
    int N, int K,
    unsigned short* __restrict__ outB)
{
  constexpr int LDK = 40;
  __shared__ __align__(16) unsigned short sA [2][64 * LDK];
  __shared__ __align__(16) unsigned short sB1[2][64 * LDK];
  __shared__ __align__(16) unsigned short sB3[2][64 * LDK];

  const int tid = threadIdx.x;
  const int m0 = blockIdx.x * 64;
  const int n0 = blockIdx.y * 64;
  const int wave = tid >> 5, lane = tid & 31;
  const int tm = (wave >> 1) * 16, tn = (wave & 1) * 32;
  const int hl = lane >> 4, ln = lane & 15;

  v8f a1L = {}, a1H = {}, a3L = {}, a3H = {};
  const int arow = tid >> 2, acol = (tid & 3) * 8;

  auto stage = [&](int buf, int k0) {
    async_ld16(&sA [buf][arow * LDK + acol], &A [(long)(m0 + arow) * K + k0 + acol]);
    async_ld16(&sB1[buf][arow * LDK + acol], &B1[(long)(n0 + arow) * K + k0 + acol]);
    async_ld16(&sB3[buf][arow * LDK + acol], &B3[(long)(n0 + arow) * K + k0 + acol]);
  };

  stage(0, 0);
  const int nk = K >> 5;
  for (int s = 0; s < nk; ++s) {
    wait_async0();
    __syncthreads();
    if (s + 1 < nk) stage((s + 1) & 1, (s + 1) << 5);
    const unsigned short* bA  = sA [s & 1];
    const unsigned short* bBa = sB1[s & 1];
    const unsigned short* bBb = sB3[s & 1];
    v16bf a   = load_frag(&bA [(tm + ln) * LDK], hl);
    v16bf b1L = load_frag(&bBa[(tn + ln) * LDK], hl);
    v16bf b1H = load_frag(&bBa[(tn + 16 + ln) * LDK], hl);
    v16bf b3L = load_frag(&bBb[(tn + ln) * LDK], hl);
    v16bf b3H = load_frag(&bBb[(tn + 16 + ln) * LDK], hl);
    a1L = __builtin_amdgcn_wmma_f32_16x16x32_bf16(false, a, false, b1L, (short)0, a1L, false, false);
    a1H = __builtin_amdgcn_wmma_f32_16x16x32_bf16(false, a, false, b1H, (short)0, a1H, false, false);
    a3L = __builtin_amdgcn_wmma_f32_16x16x32_bf16(false, a, false, b3L, (short)0, a3L, false, false);
    a3H = __builtin_amdgcn_wmma_f32_16x16x32_bf16(false, a, false, b3H, (short)0, a3H, false, false);
  }

#pragma unroll
  for (int t = 0; t < 2; ++t) {
    const v8f g1 = t ? a1H : a1L;
    const v8f g3 = t ? a3H : a3L;
#pragma unroll
    for (int r = 0; r < 8; ++r) {
      const int gm = m0 + tm + r + hl * 8;
      const int gn = n0 + tn + t * 16 + ln;
      const float h1 = g1[r], h3 = g3[r];
      outB[(long)gm * N + gn] = f2bf_u((h1 / (1.f + __expf(-h1))) * h3);
    }
  }
}

// ---------------- RMS norm: one wave (32 lanes) per token, D=128 -----------
__global__ __launch_bounds__(256) void rmsnorm_kernel(
    const float* __restrict__ x, const float* __restrict__ w,
    unsigned short* __restrict__ outB, float* __restrict__ outF)
{
  const int wid = threadIdx.x >> 5, lane = threadIdx.x & 31;
  const long tok = (long)blockIdx.x * 8 + wid;
  const long base = tok * DM + lane * 4;
  float a0 = x[base], a1 = x[base + 1], a2 = x[base + 2], a3 = x[base + 3];
  float s = a0 * a0 + a1 * a1 + a2 * a2 + a3 * a3;
#pragma unroll
  for (int o = 16; o; o >>= 1) s += __shfl_xor(s, o, 32);
  const float r = rsqrtf(s * (1.f / DM) + 1e-6f);
  const float y0 = a0 * r * w[lane * 4 + 0];
  const float y1 = a1 * r * w[lane * 4 + 1];
  const float y2 = a2 * r * w[lane * 4 + 2];
  const float y3 = a3 * r * w[lane * 4 + 3];
  if (outF) { outF[base] = y0; outF[base + 1] = y1; outF[base + 2] = y2; outF[base + 3] = y3; }
  if (outB) {
    outB[base]     = f2bf_u(y0); outB[base + 1] = f2bf_u(y1);
    outB[base + 2] = f2bf_u(y2); outB[base + 3] = f2bf_u(y3);
  }
}

// ------------- temporal attention: one block per (b, node) -----------------
__global__ __launch_bounds__(256) void attention_kernel(
    const float* __restrict__ q, const float* __restrict__ k,
    const float* __restrict__ v, float* __restrict__ out)
{
  __shared__ float sq[TSTEPS * DM], sk[TSTEPS * DM], sv[TSTEPS * DM];
  __shared__ float sp[NHEADS * TSTEPS * TSTEPS];
  const int bn = blockIdx.x;
  const int b = bn / NNODES, n = bn % NNODES;
  const int tid = threadIdx.x;
  for (int i = tid; i < TSTEPS * DM; i += 256) {
    const int t = i >> 7, d = i & 127;
    const long g = (((long)b * TSTEPS + t) * NNODES + n) * DM + d;
    sq[i] = q[g]; sk[i] = k[g]; sv[i] = v[g];
  }
  __syncthreads();
  for (int s = tid; s < NHEADS * TSTEPS * TSTEPS; s += 256) {
    const int h = s / (TSTEPS * TSTEPS), rem = s % (TSTEPS * TSTEPS);
    const int tq = rem / TSTEPS, tk = rem % TSTEPS;
    const float* qp = &sq[tq * DM + h * HDIM];
    const float* kp = &sk[tk * DM + h * HDIM];
    float a = 0.f;
#pragma unroll
    for (int d = 0; d < HDIM; ++d) a += qp[d] * kp[d];
    sp[s] = a * 0.17677669529663687f;   // 1/sqrt(32)
  }
  __syncthreads();
  if (tid < NHEADS * TSTEPS) {
    float* row = &sp[tid * TSTEPS];
    float mx = row[0];
    for (int j = 1; j < TSTEPS; ++j) mx = fmaxf(mx, row[j]);
    float den = 0.f;
    for (int j = 0; j < TSTEPS; ++j) { const float e = __expf(row[j] - mx); row[j] = e; den += e; }
    const float inv = 1.f / den;
    for (int j = 0; j < TSTEPS; ++j) row[j] *= inv;
  }
  __syncthreads();
  for (int i = tid; i < TSTEPS * DM; i += 256) {
    const int t = i >> 7, d = i & 127, h = d >> 5;
    const float* pr = &sp[(h * TSTEPS + t) * TSTEPS];
    float a = 0.f;
    for (int j = 0; j < TSTEPS; ++j) a += pr[j] * sv[j * DM + d];
    out[(((long)b * TSTEPS + t) * NNODES + n) * DM + d] = a;
  }
}

// ------------- adjacency row-normalize -> bf16, one block per row ----------
__global__ __launch_bounds__(256) void adjnorm_kernel(
    const float* __restrict__ adj, unsigned short* __restrict__ out)
{
  __shared__ float red[256];
  const int row = blockIdx.x, tid = threadIdx.x;
  const float* rp = &adj[(long)row * NNODES];
  float s = 0.f;
  for (int j = tid; j < NNODES; j += 256) s += rp[j];
  red[tid] = s; __syncthreads();
  for (int o = 128; o; o >>= 1) { if (tid < o) red[tid] += red[tid + o]; __syncthreads(); }
  const float inv = 1.f / red[0];
  for (int j = tid; j < NNODES; j += 256)
    out[(long)row * NNODES + j] = f2bf_u(rp[j] * inv);
}

// ------------- router: softmax + top-2 + shared sigmoid gate ---------------
__global__ __launch_bounds__(256) void router_kernel(
    const float* __restrict__ x2f, const float* __restrict__ gw,
    const float* __restrict__ sgw, float* __restrict__ logits_out,
    float* __restrict__ coef, float* __restrict__ sgate)
{
  const int wid = threadIdx.x >> 5, lane = threadIdx.x & 31;
  const long tok = (long)blockIdx.x * 8 + wid;
  float xa[4];
#pragma unroll
  for (int j = 0; j < 4; ++j) xa[j] = x2f[tok * DM + lane * 4 + j];
  float lg[NEXP];
#pragma unroll
  for (int e = 0; e < NEXP; ++e) {
    float s = 0.f;
#pragma unroll
    for (int j = 0; j < 4; ++j) s += xa[j] * gw[e * DM + lane * 4 + j];
#pragma unroll
    for (int o = 16; o; o >>= 1) s += __shfl_xor(s, o, 32);
    lg[e] = s;
  }
  float sg = 0.f;
#pragma unroll
  for (int j = 0; j < 4; ++j) sg += xa[j] * sgw[lane * 4 + j];
#pragma unroll
  for (int o = 16; o; o >>= 1) sg += __shfl_xor(sg, o, 32);

  if (lane == 0) {
    float mx = lg[0];
    for (int e = 1; e < NEXP; ++e) mx = fmaxf(mx, lg[e]);
    float p[NEXP]; float den = 0.f;
    for (int e = 0; e < NEXP; ++e) { p[e] = __expf(lg[e] - mx); den += p[e]; }
    int i1 = 0;
    for (int e = 1; e < NEXP; ++e) if (p[e] > p[i1]) i1 = e;
    int i2 = (i1 == 0) ? 1 : 0;
    for (int e = 0; e < NEXP; ++e) if (e != i1 && p[e] > p[i2]) i2 = e;
    const float inv = 1.f / den;
    for (int e = 0; e < NEXP; ++e) {
      logits_out[tok * NEXP + e] = lg[e];
      coef[tok * NEXP + e] = (e == i1) ? p[i1] * inv : ((e == i2) ? p[i2] * inv : 0.f);
    }
    sgate[tok] = 1.f / (1.f + __expf(-sg));
  }
}

__global__ void f2bf_kernel(const float* __restrict__ in,
                            unsigned short* __restrict__ out, long n) {
  const long i = (long)blockIdx.x * 256 + threadIdx.x;
  if (i < n) out[i] = f2bf_u(in[i]);
}

__global__ void addout_kernel(const float* __restrict__ a,
                              const float* __restrict__ b,
                              float* __restrict__ out, long n) {
  const long i = (long)blockIdx.x * 256 + threadIdx.x;
  if (i < n) out[i] = a[i] + b[i];
}

// ===========================================================================
extern "C" void kernel_launch(void* const* d_in, const int* in_sizes, int n_in,
                              void* d_out, int out_size, void* d_ws, size_t ws_size,
                              hipStream_t stream) {
  (void)in_sizes; (void)n_in; (void)out_size; (void)ws_size;
  const float* x    = (const float*)d_in[0];
  const float* kin  = (const float*)d_in[1];
  // d_in[2] (v) is unused by the reference (module passes k for both k and v)
  const float* adj  = (const float*)d_in[3];
  const float* Wq   = (const float*)d_in[4];  const float* bq   = (const float*)d_in[5];
  const float* Wk   = (const float*)d_in[6];  const float* bk   = (const float*)d_in[7];
  const float* Wv   = (const float*)d_in[8];  const float* bv   = (const float*)d_in[9];
  const float* ntw  = (const float*)d_in[10];
  const float* fc1w = (const float*)d_in[11];
  const float* mlpw = (const float*)d_in[12]; const float* mlpb = (const float*)d_in[13];
  const float* n1w  = (const float*)d_in[14];
  const float* n2w  = (const float*)d_in[15];
  const float* gatew= (const float*)d_in[16];
  const float* ew1  = (const float*)d_in[17];
  const float* ew2  = (const float*)d_in[18];
  const float* ew3  = (const float*)d_in[19];
  const float* sw1  = (const float*)d_in[20];
  const float* sw2  = (const float*)d_in[21];
  const float* sw3  = (const float*)d_in[22];
  const float* sgw  = (const float*)d_in[23];

  float* out_x      = (float*)d_out;               // [Tok, 128]
  float* out_logits = out_x + TOKS * DM;           // [Tok, 8]

  // ----- workspace arena -----
  const size_t BFB  = (size_t)TOKS * DM * 2;       // bf16 Tok x 128
  const size_t F32B = (size_t)TOKS * DM * 4;       // fp32 Tok x 128
  size_t o = 0;
  auto take = [&](size_t b) { size_t r = o; o += (b + 255) & ~(size_t)255; return r; };
  const size_t o_xnbf  = take(BFB);                 // x_n bf16; later t_x_n bf16
  const size_t o_knbf  = take(BFB);                 // k_n bf16; later mixed bf16
  const size_t o_q     = take(F32B);                // q fp32;  later shared-g bf16 (spans q+k)
  const size_t o_k     = take(F32B);
  const size_t o_v     = take(F32B);                // v fp32;  later expert-g bf16
  const size_t o_tx    = take(F32B);                // raw attn out; later x2 fp32
  const size_t o_txnf  = take(F32B);                // normalized t_x fp32
  const size_t o_gmix  = take(BFB);                 // alpha-mixed GCN hidden bf16
  const size_t o_xnew  = take(F32B);                // x + ho
  const size_t o_x2bf  = take(BFB);
  const size_t o_coef  = take((size_t)TOKS * NEXP * 4);
  const size_t o_sg    = take((size_t)TOKS * 4);
  const size_t o_final = take(F32B);
  const size_t o_adjbf = take((size_t)NNODES * NNODES * 2);
  const size_t o_wbf   = take((size_t)(16384 * 5 + 262144 * 3 + 65536 * 3) * 2);

  char* ws = (char*)d_ws;
  unsigned short* xnbf  = (unsigned short*)(ws + o_xnbf);
  unsigned short* knbf  = (unsigned short*)(ws + o_knbf);
  float*          qf    = (float*)(ws + o_q);
  float*          kf    = (float*)(ws + o_k);
  float*          vf    = (float*)(ws + o_v);
  float*          txf   = (float*)(ws + o_tx);
  float*          txnf  = (float*)(ws + o_txnf);
  unsigned short* gmixb = (unsigned short*)(ws + o_gmix);
  float*          xnewf = (float*)(ws + o_xnew);
  unsigned short* x2bf  = (unsigned short*)(ws + o_x2bf);
  float*          coeff = (float*)(ws + o_coef);
  float*          sgf   = (float*)(ws + o_sg);
  float*          finalf= (float*)(ws + o_final);
  unsigned short* adjbf = (unsigned short*)(ws + o_adjbf);
  unsigned short* wbf   = (unsigned short*)(ws + o_wbf);
  // aliases (regions dead by the time they are reused)
  unsigned short* txnbf = xnbf;
  unsigned short* mixbf = knbf;
  unsigned short* gexpb = (unsigned short*)(ws + o_v);   // Tok x 256 bf16
  unsigned short* gshb  = (unsigned short*)(ws + o_q);   // Tok x 512 bf16 (spans q+k)
  float*          x2f   = txf;

  // bf16 weight sub-offsets (in shorts)
  unsigned short* wq_b  = wbf;
  unsigned short* wk_b  = wbf + 16384;
  unsigned short* wv_b  = wbf + 32768;
  unsigned short* fc1_b = wbf + 49152;
  unsigned short* mlp_b = wbf + 65536;
  unsigned short* ew1_b = wbf + 81920;    // 8 x 256 x 128
  unsigned short* ew2_b = wbf + 344064;   // 8 x 128 x 256
  unsigned short* ew3_b = wbf + 606208;
  unsigned short* sw1_b = wbf + 868352;   // 512 x 128
  unsigned short* sw2_b = wbf + 933888;   // 128 x 512
  unsigned short* sw3_b = wbf + 999424;

  auto conv = [&](const float* src, unsigned short* dst, long n) {
    f2bf_kernel<<<dim3((unsigned)((n + 255) / 256)), 256, 0, stream>>>(src, dst, n);
  };

  // 1) weights -> bf16
  conv(Wq, wq_b, 16384);   conv(Wk, wk_b, 16384);   conv(Wv, wv_b, 16384);
  conv(fc1w, fc1_b, 16384); conv(mlpw, mlp_b, 16384);
  conv(ew1, ew1_b, 262144); conv(ew2, ew2_b, 262144); conv(ew3, ew3_b, 262144);
  conv(sw1, sw1_b, 65536);  conv(sw2, sw2_b, 65536);  conv(sw3, sw3_b, 65536);

  // 2) row-normalized adjacency -> bf16
  adjnorm_kernel<<<NNODES, 256, 0, stream>>>(adj, adjbf);

  // 3) pre-norm
  rmsnorm_kernel<<<TOKS / 8, 256, 0, stream>>>(x,   n1w, xnbf, nullptr);
  rmsnorm_kernel<<<TOKS / 8, 256, 0, stream>>>(kin, n1w, knbf, nullptr);

  // 4) QKV projections (bf16 WMMA, fp32 out)
  const dim3 gTok((unsigned)(TOKS / 64), DM / 64, 1);
  gemm_wmma<0, false><<<gTok, 256, 0, stream>>>(xnbf, wq_b, (int)TOKS, DM, DM,
      0, 0, 0, qf, nullptr, bq, nullptr, nullptr, 0, 0.f);
  gemm_wmma<0, false><<<gTok, 256, 0, stream>>>(knbf, wk_b, (int)TOKS, DM, DM,
      0, 0, 0, kf, nullptr, bk, nullptr, nullptr, 0, 0.f);
  gemm_wmma<0, false><<<gTok, 256, 0, stream>>>(knbf, wv_b, (int)TOKS, DM, DM,
      0, 0, 0, vf, nullptr, bv, nullptr, nullptr, 0, 0.f);

  // 5) temporal attention
  attention_kernel<<<4 * NNODES, 256, 0, stream>>>(qf, kf, vf, txf);

  // 6) norm_tatt -> bf16 (GEMM input) + fp32 (residual mix)
  rmsnorm_kernel<<<TOKS / 8, 256, 0, stream>>>(txf, ntw, txnbf, txnf);

  // 7) GCN neighbor mix: mixed[b,t] = a_norm @ t_x_n[b,t]  (96 slices)
  gemm_wmma<1, true><<<dim3(NNODES / 64, DM / 64, 96), 256, 0, stream>>>(
      adjbf, txnbf, NNODES, DM, NNODES,
      0, (long)NNODES * DM, (long)NNODES * DM,
      nullptr, mixbf, nullptr, nullptr, nullptr, 0, 0.f);

  // 8) gcn_fc1 + alpha residual mix -> bf16
  gemm_wmma<2, false><<<gTok, 256, 0, stream>>>(mixbf, fc1_b, (int)TOKS, DM, DM,
      0, 0, 0, nullptr, gmixb, nullptr, txnf, nullptr, 0, ALPHA_MIX);

  // 9) gcn_mlp + bias + block residual -> x_new fp32
  gemm_wmma<3, false><<<gTok, 256, 0, stream>>>(gmixb, mlp_b, (int)TOKS, DM, DM,
      0, 0, 0, xnewf, nullptr, mlpb, x, nullptr, 0, 0.f);

  // 10) norm2 -> bf16 (expert GEMMs) + fp32 (router)
  rmsnorm_kernel<<<TOKS / 8, 256, 0, stream>>>(xnewf, n2w, x2bf, x2f);

  // 11) router: logits (output 2), top-2 coef, shared sigmoid gate
  router_kernel<<<TOKS / 8, 256, 0, stream>>>(x2f, gatew, sgw,
                                              out_logits, coeff, sgf);

  // 12) MoE accumulator = 0
  hipMemsetAsync(finalf, 0, F32B, stream);

  // 13) experts (dense like the reference; coef zero for non-top-2)
  for (int e = 0; e < NEXP; ++e) {
    gemm_wmma_swiglu<<<dim3((unsigned)(TOKS / 64), MINTER / 64, 1), 256, 0, stream>>>(
        x2bf, ew1_b + (long)e * MINTER * DM, ew3_b + (long)e * MINTER * DM,
        MINTER, DM, gexpb);
    gemm_wmma<4, false><<<gTok, 256, 0, stream>>>(gexpb, ew2_b + (long)e * DM * MINTER,
        (int)TOKS, DM, MINTER, 0, 0, 0,
        finalf, nullptr, nullptr, nullptr, coeff + e, NEXP, 0.f);
  }

  // 14) shared expert, gated by sigmoid(x2 @ sgate^T)
  gemm_wmma_swiglu<<<dim3((unsigned)(TOKS / 64), INTERD / 64, 1), 256, 0, stream>>>(
      x2bf, sw1_b, sw3_b, INTERD, DM, gshb);
  gemm_wmma<4, false><<<gTok, 256, 0, stream>>>(gshb, sw2_b,
      (int)TOKS, DM, INTERD, 0, 0, 0,
      finalf, nullptr, nullptr, nullptr, sgf, 1, 0.f);

  // 15) output: x_new + final
  addout_kernel<<<dim3((unsigned)((TOKS * DM) / 256)), 256, 0, stream>>>(
      xnewf, finalf, out_x, TOKS * DM);
}
